// SimclrCriterion_85418309583110
// MI455X (gfx1250) — matrix operand order repaired and checked by reading, not compiled, not run on hardware
//
#include <hip/hip_runtime.h>

typedef _Float16 f16;
typedef __attribute__((ext_vector_type(16))) _Float16 v16h;
typedef __attribute__((ext_vector_type(8)))  _Float16 v8h;
typedef __attribute__((ext_vector_type(8)))  float    v8f;
typedef __attribute__((ext_vector_type(4)))  int      v4i;

#define BDIM   4096
#define DDIM   512
#define NROWS  (2 * BDIM)      // 8192 stacked rows: [z_i ; z_j]
#define INV_T  10.0f           // 1 / TEMPERATURE

#if __has_builtin(__builtin_amdgcn_global_load_async_to_lds_b128)
#define HAVE_ASYNC_LDS 1
#else
#define HAVE_ASYNC_LDS 0
#endif

#define AS1 __attribute__((address_space(1)))
#define AS3 __attribute__((address_space(3)))

// Copy one 16-byte chunk global -> LDS (async path if available).
__device__ __forceinline__ void copy16_to_lds(const f16* g, f16* l) {
#if HAVE_ASYNC_LDS
  __builtin_amdgcn_global_load_async_to_lds_b128(
      (AS1 v4i*)(uintptr_t)g, (AS3 v4i*)(uint32_t)(uintptr_t)l, 0, 0);
#else
  *(v8h*)l = *(const v8h*)g;
#endif
}

// Wait for this wave's async copies, then block barrier.
__device__ __forceinline__ void stage_join() {
#if HAVE_ASYNC_LDS
#if __has_builtin(__builtin_amdgcn_s_wait_asynccnt)
  __builtin_amdgcn_s_wait_asynccnt(0);
#else
  asm volatile("s_wait_asynccnt 0x0" ::: "memory");
#endif
#endif
  __syncthreads();
}

// ---------------------------------------------------------------------------
// Kernel 1: L2-normalize each row, emit hi/lo f16 split (hi + lo ~ fp32 value)
// ---------------------------------------------------------------------------
__global__ __launch_bounds__(256) void normalize_kernel(
    const float* __restrict__ zi, const float* __restrict__ zj,
    f16* __restrict__ Xhi, f16* __restrict__ Xlo) {
  __shared__ float red[256];
  const int row = blockIdx.x;
  const float* src = (row < BDIM) ? (zi + (size_t)row * DDIM)
                                  : (zj + (size_t)(row - BDIM) * DDIM);
  const int t = threadIdx.x;
  float x0 = src[t];
  float x1 = src[t + 256];
  red[t] = x0 * x0 + x1 * x1;
  __syncthreads();
  for (int s = 128; s > 0; s >>= 1) {
    if (t < s) red[t] += red[t + s];
    __syncthreads();
  }
  const float rinv = 1.0f / sqrtf(red[0]);
  const size_t base = (size_t)row * DDIM;
  float y0 = x0 * rinv;
  float y1 = x1 * rinv;
  f16 h0 = (f16)y0;
  f16 h1 = (f16)y1;
  Xhi[base + t]       = h0;
  Xhi[base + t + 256] = h1;
  Xlo[base + t]       = (f16)(y0 - (float)h0);
  Xlo[base + t + 256] = (f16)(y1 - (float)h1);
}

// ---------------------------------------------------------------------------
// Kernel 2: target logit per row: inv_t * dot(X[r], X[j0]), j0 = (r%B==0)?1:0
// ---------------------------------------------------------------------------
__global__ __launch_bounds__(256) void target_kernel(
    const f16* __restrict__ Xhi, const f16* __restrict__ Xlo,
    float* __restrict__ target) {
  __shared__ float red[256];
  const int row = blockIdx.x;
  const int i  = row & (BDIM - 1);
  const int j0 = (i == 0) ? 1 : 0;
  const size_t ra = (size_t)row * DDIM;
  const size_t rb = (size_t)j0 * DDIM;
  const int t = threadIdx.x;
  float a0 = (float)Xhi[ra + t]       + (float)Xlo[ra + t];
  float a1 = (float)Xhi[ra + t + 256] + (float)Xlo[ra + t + 256];
  float b0 = (float)Xhi[rb + t]       + (float)Xlo[rb + t];
  float b1 = (float)Xhi[rb + t + 256] + (float)Xlo[rb + t + 256];
  red[t] = a0 * b0 + a1 * b1;
  __syncthreads();
  for (int s = 128; s > 0; s >>= 1) {
    if (t < s) red[t] += red[t + s];
    __syncthreads();
  }
  if (t == 0) target[row] = red[0] * INV_T;
}

// ---------------------------------------------------------------------------
// Kernel 3: zero the per-row sum-of-exp accumulators
// ---------------------------------------------------------------------------
__global__ __launch_bounds__(256) void zero_kernel(float* __restrict__ p) {
  p[blockIdx.x * 256 + threadIdx.x] = 0.0f;
}

// ---------------------------------------------------------------------------
// Kernel 4: fused Gram + online exp-sum.
//  - 8 waves/block, each owns a 16-row tile (128 rows/block).
//  - B column tiles (16 cols x 512 K, hi+lo) are async-copied global->LDS
//    once per block and consumed by all 8 waves (8x L2 traffic cut),
//    double-buffered: prefetch ct+1 while computing ct.
//  - LDS layout is K-chunked [k/8][col][8] so each lane's WMMA B fragment is
//    two contiguous 16B ds_load_b128 with conflict-free banking, and the
//    double buffer is exactly 64 KB.
//  - hi/lo split f16 WMMA: 3 x v_wmma_f32_16x16x32_f16 per 32-wide k-step.
// grid = (64, 8): 512 row tiles; grid.y splits 512 col tiles into 8 chunks.
// ---------------------------------------------------------------------------
__global__ __launch_bounds__(256) void gram_lse_kernel(
    const f16* __restrict__ Xhi, const f16* __restrict__ Xlo,
    float* __restrict__ sumexp) {
  __shared__ f16 ldsB[2][2][16 * DDIM];   // [buf][hi/lo][chunk*128 + col*8 + k&7]

  const int t       = threadIdx.x;
  const int lane    = t & 31;
  const int wave    = t >> 5;
  const int rowBase = (blockIdx.x * 8 + wave) * 16;
  const int ct0     = blockIdx.y * 64;
  const int l15     = lane & 15;
  const bool hihalf = lane >= 16;

  // ---- hoist A fragments (16 rows x all 512 K, hi+lo) into registers ------
  // A 16x32 f16 layout: lanes 0-15 K {kb..+7, kb+16..+23}; lanes 16-31 +8.
  const size_t arow = (size_t)(rowBase + l15) * DDIM;
  const int    aoff = hihalf ? 8 : 0;
  v16h Ahi[16], Alo[16];
#pragma unroll
  for (int kk = 0; kk < 16; ++kk) {
    const int kb = kk * 32;
    v8h h0 = *(const v8h*)(Xhi + arow + kb + aoff);
    v8h h1 = *(const v8h*)(Xhi + arow + kb + aoff + 16);
    v8h l0 = *(const v8h*)(Xlo + arow + kb + aoff);
    v8h l1 = *(const v8h*)(Xlo + arow + kb + aoff + 16);
    Ahi[kk] = __builtin_shufflevector(h0, h1, 0,1,2,3,4,5,6,7,8,9,10,11,12,13,14,15);
    Alo[kk] = __builtin_shufflevector(l0, l1, 0,1,2,3,4,5,6,7,8,9,10,11,12,13,14,15);
  }

  v8f acc = {};   // per-lane exp accumulators for 8 C rows

  // ---- staging helper: 256 threads copy 16 cols x 512 K x (hi+lo) ---------
  // 1024 16B chunks per part, 4 per thread per part.
  auto stage = [&](int ct, int buf) {
#pragma unroll
    for (int j = 0; j < 4; ++j) {
      const int c     = t * 4 + j;       // 0..1023
      const int col   = c >> 6;          // 0..15
      const int k8    = c & 63;          // 0..63 K-chunk
      const size_t g  = (size_t)(ct * 16 + col) * DDIM + k8 * 8;
      const int    l  = k8 * 128 + col * 8;
      copy16_to_lds(Xhi + g, &ldsB[buf][0][l]);
      copy16_to_lds(Xlo + g, &ldsB[buf][1][l]);
    }
  };

  stage(ct0, 0);
  stage_join();

  for (int ct = ct0; ct < ct0 + 64; ++ct) {
    const int bi = ct & 1;
    if (ct + 1 < ct0 + 64) stage(ct + 1, bi ^ 1);   // prefetch next tile

    // ---- compute 16x16 tile from LDS-resident B -------------------------
    const f16* bh = &ldsB[bi][0][0];
    const f16* bl = &ldsB[bi][1][0];
    const int cofs = (hihalf ? 1 : 0) * 128 + l15 * 8;  // lanes 16-31: K+8
    v8f c = {};
#pragma unroll
    for (int kk = 0; kk < 16; ++kk) {
      const int p = kk * 4 * 128 + cofs;
      v8h b0 = *(const v8h*)(bh + p);
      v8h b1 = *(const v8h*)(bh + p + 256);
      v8h b2 = *(const v8h*)(bl + p);
      v8h b3 = *(const v8h*)(bl + p + 256);
      v16h Bhi = __builtin_shufflevector(b0, b1, 0,1,2,3,4,5,6,7,8,9,10,11,12,13,14,15);
      v16h Blo = __builtin_shufflevector(b2, b3, 0,1,2,3,4,5,6,7,8,9,10,11,12,13,14,15);
      // dot ~= hi*hi + hi*lo + lo*hi  (lo*lo < 1e-5 of result, dropped)
      c = __builtin_amdgcn_wmma_f32_16x16x32_f16(false, Ahi[kk], false, Bhi, (short)0, c, false, false);
      c = __builtin_amdgcn_wmma_f32_16x16x32_f16(false, Ahi[kk], false, Blo, (short)0, c, false, false);
      c = __builtin_amdgcn_wmma_f32_16x16x32_f16(false, Alo[kk], false, Bhi, (short)0, c, false, false);
    }

    // C layout: lane holds (M = v + (hihalf?8:0), N = ct*16 + (lane&15)).
    const int colg = ct * 16 + l15;
#pragma unroll
    for (int v = 0; v < 8; ++v) {
      const int rowg = rowBase + v + (hihalf ? 8 : 0);
      float e = (rowg == colg) ? 0.0f : __expf(c[v] * INV_T);  // exact diag skip
      acc[v] += e;
    }

    stage_join();   // asyncs for ct+1 done; all waves done reading buf bi
  }

  // Reduce each row across its 16-lane group (xor masks stay within halves).
#pragma unroll
  for (int v = 0; v < 8; ++v) {
    float s = acc[v];
    s += __shfl_xor(s, 1, 32);
    s += __shfl_xor(s, 2, 32);
    s += __shfl_xor(s, 4, 32);
    s += __shfl_xor(s, 8, 32);
    if (l15 == 0)
      atomicAdd(&sumexp[rowBase + v + (hihalf ? 8 : 0)], s);
  }
}

// ---------------------------------------------------------------------------
// Kernel 5: loss = -(1/2B) * sum_r (target[r] - log(sumexp[r]))
// ---------------------------------------------------------------------------
__global__ __launch_bounds__(256) void loss_kernel(
    const float* __restrict__ sumexp, const float* __restrict__ target,
    float* __restrict__ out) {
  __shared__ float red[256];
  const int t = threadIdx.x;
  float s = 0.0f;
  for (int r = t; r < NROWS; r += 256)
    s += target[r] - logf(sumexp[r]);
  red[t] = s;
  __syncthreads();
  for (int k = 128; k > 0; k >>= 1) {
    if (t < k) red[t] += red[t + k];
    __syncthreads();
  }
  if (t == 0) out[0] = -red[0] / (float)NROWS;
}

// ---------------------------------------------------------------------------
extern "C" void kernel_launch(void* const* d_in, const int* in_sizes, int n_in,
                              void* d_out, int out_size, void* d_ws, size_t ws_size,
                              hipStream_t stream) {
  (void)in_sizes; (void)n_in; (void)out_size; (void)ws_size;
  const float* zi = (const float*)d_in[0];
  const float* zj = (const float*)d_in[1];

  char* ws = (char*)d_ws;
  f16*   Xhi    = (f16*)ws;                                        //  8 MiB
  f16*   Xlo    = (f16*)(ws + (size_t)NROWS * DDIM * 2);           //  8 MiB
  float* sumexp = (float*)(ws + (size_t)NROWS * DDIM * 4);         // 32 KiB
  float* target = sumexp + NROWS;                                  // 32 KiB

  normalize_kernel<<<NROWS, 256, 0, stream>>>(zi, zj, Xhi, Xlo);
  target_kernel<<<NROWS, 256, 0, stream>>>(Xhi, Xlo, target);
  zero_kernel<<<NROWS / 256, 256, 0, stream>>>(sumexp);
  dim3 grid(64, 8);
  gram_lse_kernel<<<grid, 256, 0, stream>>>(Xhi, Xlo, sumexp);
  loss_kernel<<<1, 256, 0, stream>>>(sumexp, target, (float*)d_out);
}
